// SNNLanguageModel_36275293782719
// MI455X (gfx1250) — matrix operand order repaired and checked by reading, not compiled, not run on hardware
//
#include <hip/hip_runtime.h>
#include <hip/hip_bf16.h>

// ---------------- model constants ----------------
#define BB   4      // batch
#define SS   128    // seq
#define DD   1024   // model dim
#define NST  4      // streams
#define KK   4      // frames per token
#define LL   3      // layers
#define DFF  3072
#define NDD  4096   // N*D
#define VV   6144   // vocab
#define TT   (SS*KK)  // 512 frames

typedef __attribute__((ext_vector_type(16))) __bf16 v16bf;
typedef __attribute__((ext_vector_type(8)))  float  v8f;

union FragBf {
    v16bf v;
    uint4 q[2];
};

__device__ __forceinline__ unsigned short f2bf(float f) {
    union { float f; unsigned int u; } c; c.f = f;
    unsigned int u = c.u + 0x7FFFu + ((c.u >> 16) & 1u);   // RNE
    return (unsigned short)(u >> 16);
}

// ---------------- WMMA GEMM ----------------
// C[M,N](f32) = A[M,K](bf16 row-major) x Bt[N,K](bf16, pre-transposed)^T
// EPI: 0 = none, 1 = += aux[row,col] (addsrc), 2 = += aux[col] (bias)

__device__ __forceinline__ void loadA_frag(FragBf& f, const unsigned short* __restrict__ A,
                                           int row, int K, int kb, int half) {
    const unsigned short* base = A + (size_t)row * K + kb + 8 * half;
    f.q[0] = *(const uint4*)base;          // K chunk0: 8 bf16
    f.q[1] = *(const uint4*)(base + 16);   // K chunk1: 8 bf16
}

__device__ __forceinline__ void loadB_frag(FragBf& f, const unsigned short* __restrict__ Bt,
                                           int col, int K, int kb, int half) {
    const uint4* p = (const uint4*)(Bt + (size_t)col * K + kb + 8 * half);
    f.q[0] = p[0];
    f.q[1] = p[2];
}

template <int EPI>
__global__ void __launch_bounds__(256)
gemm_bf16_kernel(const unsigned short* __restrict__ A, const unsigned short* __restrict__ Bt,
                 float* __restrict__ C, const float* __restrict__ aux,
                 int M, int N, int K) {
    int lane = threadIdx.x & 31;
    int wid  = (blockIdx.x * blockDim.x + threadIdx.x) >> 5;
    int wavesN = N >> 6;   // 64 cols per wave
    int wavesM = M >> 5;   // 32 rows per wave
    if (wid >= wavesM * wavesN) return;
    int wm = wid / wavesN, wn = wid % wavesN;
    int rowBase = wm << 5, colBase = wn << 6;
    int half = lane >> 4, r = lane & 15;

    v8f acc[2][4] = {};
    for (int kb = 0; kb < K; kb += 32) {
        FragBf a0, a1, b0, b1, b2, b3;
        loadA_frag(a0, A, rowBase + r,      K, kb, half);
        loadA_frag(a1, A, rowBase + 16 + r, K, kb, half);
        loadB_frag(b0, Bt, colBase + r,      K, kb, half);
        loadB_frag(b1, Bt, colBase + 16 + r, K, kb, half);
        loadB_frag(b2, Bt, colBase + 32 + r, K, kb, half);
        loadB_frag(b3, Bt, colBase + 48 + r, K, kb, half);
        // prefetch next K-block of B (gfx1250 global_prefetch path)
        __builtin_prefetch(Bt + (size_t)(colBase + r) * K + kb + 32 + 16 * half, 0, 0);

        acc[0][0] = __builtin_amdgcn_wmma_f32_16x16x32_bf16(false, a0.v, false, b0.v, (short)0, acc[0][0], false, false);
        acc[0][1] = __builtin_amdgcn_wmma_f32_16x16x32_bf16(false, a0.v, false, b1.v, (short)0, acc[0][1], false, false);
        acc[0][2] = __builtin_amdgcn_wmma_f32_16x16x32_bf16(false, a0.v, false, b2.v, (short)0, acc[0][2], false, false);
        acc[0][3] = __builtin_amdgcn_wmma_f32_16x16x32_bf16(false, a0.v, false, b3.v, (short)0, acc[0][3], false, false);
        acc[1][0] = __builtin_amdgcn_wmma_f32_16x16x32_bf16(false, a1.v, false, b0.v, (short)0, acc[1][0], false, false);
        acc[1][1] = __builtin_amdgcn_wmma_f32_16x16x32_bf16(false, a1.v, false, b1.v, (short)0, acc[1][1], false, false);
        acc[1][2] = __builtin_amdgcn_wmma_f32_16x16x32_bf16(false, a1.v, false, b2.v, (short)0, acc[1][2], false, false);
        acc[1][3] = __builtin_amdgcn_wmma_f32_16x16x32_bf16(false, a1.v, false, b3.v, (short)0, acc[1][3], false, false);
    }

    #pragma unroll
    for (int i = 0; i < 2; i++) {
        #pragma unroll
        for (int j = 0; j < 4; j++) {
            int col = colBase + j * 16 + r;
            #pragma unroll
            for (int v = 0; v < 8; v++) {
                int row = rowBase + i * 16 + v + 8 * half;
                size_t idx = (size_t)row * N + col;
                float val = acc[i][j][v];
                if (EPI == 1) val += aux[idx];
                if (EPI == 2) val += aux[col];
                C[idx] = val;
            }
        }
    }
}

// ---------------- weight convert / transpose to bf16 ----------------
__global__ void __launch_bounds__(256)
convT_kernel(const float* __restrict__ W, unsigned short* __restrict__ Wt, int K, int N) {
    size_t idx = (size_t)blockIdx.x * blockDim.x + threadIdx.x;
    if (idx >= (size_t)K * N) return;
    int n = (int)(idx / K), k = (int)(idx % K);
    Wt[idx] = f2bf(W[(size_t)k * N + n]);
}

__global__ void __launch_bounds__(256)
convC_kernel(const float* __restrict__ W, unsigned short* __restrict__ O, long n) {
    long idx = (long)blockIdx.x * blockDim.x + threadIdx.x;
    if (idx >= n) return;
    O[idx] = f2bf(W[idx]);
}

// ---------------- embedding gather + stream broadcast ----------------
__global__ void __launch_bounds__(256)
embed_init_kernel(const int* __restrict__ tok, const float* __restrict__ emb,
                  float* __restrict__ h) {
    int idx = blockIdx.x * blockDim.x + threadIdx.x;  // over S*B*D
    if (idx >= SS * BB * DD) return;
    int d = idx % DD, sb = idx / DD;
    int b = sb % BB, s = sb / BB;
    float v = emb[(size_t)tok[b * SS + s] * DD + d];
    size_t base = ((size_t)sb * NST) * DD + d;
    #pragma unroll
    for (int m = 0; m < NST; m++) h[base + (size_t)m * DD] = v;
}

// ---------------- per-layer params: sinkhorn(H), softmax(w_pre/w_post), zero ek ----------------
__global__ void
params_kernel(const float* __restrict__ Hl, const float* __restrict__ wpre_in,
              const float* __restrict__ wpost_in, float* __restrict__ Hout,
              float* __restrict__ wpre_o, float* __restrict__ wpost_o,
              float* __restrict__ ek) {
    int l = blockIdx.x;
    if (threadIdx.x != 0) return;
    float m[16];
    for (int i = 0; i < 16; i++) m[i] = __expf(Hl[l * 16 + i]);
    for (int it = 0; it < 20; it++) {
        for (int r0 = 0; r0 < 4; r0++) {
            float s = m[r0*4+0] + m[r0*4+1] + m[r0*4+2] + m[r0*4+3];
            for (int c = 0; c < 4; c++) m[r0*4+c] /= s;
        }
        for (int c = 0; c < 4; c++) {
            float s = m[c] + m[4+c] + m[8+c] + m[12+c];
            for (int r0 = 0; r0 < 4; r0++) m[r0*4+c] /= s;
        }
    }
    for (int i = 0; i < 16; i++) Hout[l * 16 + i] = m[i];
    {
        float mx = -1e30f;
        for (int i = 0; i < 4; i++) mx = fmaxf(mx, wpre_in[l*4+i]);
        float e[4], s = 0.f;
        for (int i = 0; i < 4; i++) { e[i] = __expf(wpre_in[l*4+i] - mx); s += e[i]; }
        for (int i = 0; i < 4; i++) wpre_o[l*4+i] = e[i] / s;
    }
    {
        float mx = -1e30f;
        for (int i = 0; i < 4; i++) mx = fmaxf(mx, wpost_in[l*4+i]);
        float e[4], s = 0.f;
        for (int i = 0; i < 4; i++) { e[i] = __expf(wpost_in[l*4+i] - mx); s += e[i]; }
        for (int i = 0; i < 4; i++) wpost_o[l*4+i] = e[i] / s;
    }
    ek[l] = 0.f;
}

// ---------------- stream mixing + pre-collapse ----------------
__global__ void __launch_bounds__(256)
mix_kernel(const float* __restrict__ h, const float* __restrict__ Hm,
           const float* __restrict__ wpre, float* __restrict__ hmix,
           float* __restrict__ x) {
    int idx = blockIdx.x * blockDim.x + threadIdx.x;  // over S*B*D
    if (idx >= SS * BB * DD) return;
    int d = idx % DD; size_t sb = idx / DD;
    size_t base = sb * NST * DD + d;
    float h0 = h[base], h1 = h[base + DD], h2 = h[base + 2*DD], h3 = h[base + 3*DD];
    float xx = 0.f;
    #pragma unroll
    for (int m = 0; m < NST; m++) {
        float v = Hm[m*4+0]*h0 + Hm[m*4+1]*h1 + Hm[m*4+2]*h2 + Hm[m*4+3]*h3;
        hmix[base + (size_t)m * DD] = v;
        xx += wpre[m] * v;
    }
    x[sb * DD + d] = xx;
}

// ---------------- generic PLIF scan over T, bf16 output ----------------
// x1: main input ((T or S),Brows,C) -- if rep!=0 the time index is t>>2 (K-repeat)
// x2: optional second addend (T,Brows,C). Output is bf16 (feeds a WMMA GEMM).
__global__ void __launch_bounds__(256)
plif_kernel(const float* __restrict__ x1, int rep, const float* __restrict__ x2,
            const float* __restrict__ beta, const float* __restrict__ vth,
            unsigned short* __restrict__ out, int T, int Brows, int C) {
    int idx = blockIdx.x * blockDim.x + threadIdx.x;  // over Brows*C
    if (idx >= Brows * C) return;
    int c = idx % C;
    float be = beta[c], th = vth[c];
    float v = 0.f;
    size_t stride = (size_t)Brows * C;
    for (int t = 0; t < T; t++) {
        size_t o = (size_t)t * stride + idx;
        float in = rep ? x1[(size_t)(t >> 2) * stride + idx] : x1[o];
        if (x2) in += x2[o];
        v = be * v + (1.f - be) * in;
        float s = (v >= th) ? th : 0.f;
        v -= s;
        out[o] = f2bf(s);
    }
}

// ---------------- fused gate/up double PLIF scan, writes bf16 g*u ----------------
__global__ void __launch_bounds__(256)
gateup_kernel(const float* __restrict__ gin, const float* __restrict__ uin,
              const float* __restrict__ bg, const float* __restrict__ vg,
              const float* __restrict__ bu, const float* __restrict__ vu,
              unsigned short* __restrict__ out, int T, int Brows, int C) {
    int idx = blockIdx.x * blockDim.x + threadIdx.x;
    if (idx >= Brows * C) return;
    int c = idx % C;
    float beg = bg[c], thg = vg[c], beu = bu[c], thu = vu[c];
    float vgs = 0.f, vus = 0.f;
    size_t stride = (size_t)Brows * C;
    for (int t = 0; t < T; t++) {
        size_t o = (size_t)t * stride + idx;
        float g = gin[o], u = uin[o];
        vgs = beg * vgs + (1.f - beg) * g;
        float sg = (vgs >= thg) ? thg : 0.f; vgs -= sg;
        vus = beu * vus + (1.f - beu) * u;
        float su = (vus >= thu) ? thu : 0.f; vus -= su;
        out[o] = f2bf(sg * su);
    }
}

// ---------------- halting lambda: one wave per (t,b) row ----------------
__global__ void __launch_bounds__(256)
halt_lam_kernel(const float* __restrict__ y, const float* __restrict__ hw,
                const float* __restrict__ hbp, float* __restrict__ lam, int rows) {
    int wid  = (blockIdx.x * blockDim.x + threadIdx.x) >> 5;
    int lane = threadIdx.x & 31;
    if (wid >= rows) return;
    const float* yr = y + (size_t)wid * DD;
    float s = 0.f;
    for (int i = lane; i < DD; i += 32) s += yr[i] * hw[i];
    #pragma unroll
    for (int off = 16; off; off >>= 1) s += __shfl_xor(s, off, 32);
    if (lane == 0) lam[wid] = 1.f / (1.f + __expf(-(s + hbp[0])));
}

// ---------------- PonderNet pk + ek accumulation ----------------
__global__ void __launch_bounds__(256)
pk_ek_kernel(const float* __restrict__ lam, float* __restrict__ pk,
             float* __restrict__ ek_acc) {
    int idx = blockIdx.x * blockDim.x + threadIdx.x;   // over S*B
    if (idx >= SS * BB) return;
    int s = idx / BB, b = idx % BB;
    float p[KK]; float prev = 1.f, tot = 0.f;
    #pragma unroll
    for (int k = 0; k < KK; k++) {
        float l = lam[(s * KK + k) * BB + b];
        p[k] = l * prev;
        prev *= (1.f - l);
        tot += p[k];
    }
    p[KK - 1] += 1.f - tot;
    float ek = 0.f;
    #pragma unroll
    for (int k = 0; k < KK; k++) { pk[idx * KK + k] = p[k]; ek += p[k] * (float)(k + 1); }
    atomicAdd(ek_acc, ek / (float)(SS * BB));
}

// ---------------- y_tok + residual stream update ----------------
__global__ void __launch_bounds__(256)
ytok_update_kernel(const float* __restrict__ y, const float* __restrict__ pk,
                   const float* __restrict__ hmix, const float* __restrict__ wpost,
                   float* __restrict__ h) {
    int idx = blockIdx.x * blockDim.x + threadIdx.x;   // over S*B*D
    if (idx >= SS * BB * DD) return;
    int d = idx % DD; int sb = idx / DD;
    int b = sb % BB, s = sb / BB;
    float yt = 0.f;
    #pragma unroll
    for (int k = 0; k < KK; k++)
        yt += pk[sb * KK + k] * y[((size_t)(s * KK + k) * BB + b) * DD + d];
    size_t base = (size_t)sb * NST * DD + d;
    #pragma unroll
    for (int m = 0; m < NST; m++)
        h[base + (size_t)m * DD] = hmix[base + (size_t)m * DD] + wpost[m] * yt;
}

// ---------------- stream collapse + RMSNorm ----------------
__global__ void __launch_bounds__(256)
collapse_rms_kernel(const float* __restrict__ h, const float* __restrict__ w,
                    float* __restrict__ hn) {
    int sb = blockIdx.x;                         // S*B blocks
    __shared__ float red[256];
    const float* hb = h + (size_t)sb * NST * DD;
    float vals[DD / 256]; float ss = 0.f;
    #pragma unroll
    for (int i = 0; i < DD / 256; i++) {
        int d = threadIdx.x + i * 256;
        float v = 0.25f * (hb[d] + hb[DD + d] + hb[2*DD + d] + hb[3*DD + d]);
        vals[i] = v; ss += v * v;
    }
    red[threadIdx.x] = ss; __syncthreads();
    for (int o = 128; o; o >>= 1) {
        if (threadIdx.x < o) red[threadIdx.x] += red[threadIdx.x + o];
        __syncthreads();
    }
    float inv = rsqrtf(red[0] / (float)DD + 1e-6f);
    #pragma unroll
    for (int i = 0; i < DD / 256; i++) {
        int d = threadIdx.x + i * 256;
        hn[(size_t)sb * DD + d] = vals[i] * w[d] * inv;
    }
}

// ---------------- output PLIF + K-frame mean + (B,S,D) transpose, bf16 out ----------------
__global__ void __launch_bounds__(256)
plif_out_kernel(const float* __restrict__ hn, const float* __restrict__ beta,
                const float* __restrict__ vth, unsigned short* __restrict__ decoded) {
    int idx = blockIdx.x * blockDim.x + threadIdx.x;   // over B*D
    if (idx >= BB * DD) return;
    int d = idx % DD, b = idx / DD;
    float be = beta[d], th = vth[d], v = 0.f, acc = 0.f;
    for (int t = 0; t < TT; t++) {
        int s = t >> 2;
        float in = hn[((size_t)s * BB + b) * DD + d];
        v = be * v + (1.f - be) * in;
        float sp = (v >= th) ? th : 0.f;
        v -= sp; acc += sp;
        if ((t & 3) == 3) {
            decoded[((size_t)b * SS + s) * DD + d] = f2bf(0.25f * acc);
            acc = 0.f;
        }
    }
}

// ---------------- lateral inhibition: fp32 in, bf16 out ----------------
__global__ void __launch_bounds__(256)
lateral_kernel(const float* __restrict__ hh, const float* __restrict__ w,
               unsigned short* __restrict__ out16) {
    int row = blockIdx.x;                        // B*S rows
    __shared__ float rs[256], rq[256];
    float vals[DD / 256]; float s = 0.f, q = 0.f;
    #pragma unroll
    for (int i = 0; i < DD / 256; i++) {
        int d = threadIdx.x + i * 256;
        float v = hh[(size_t)row * DD + d];
        vals[i] = v; s += v; q += v * v;
    }
    rs[threadIdx.x] = s; rq[threadIdx.x] = q; __syncthreads();
    for (int o = 128; o; o >>= 1) {
        if (threadIdx.x < o) { rs[threadIdx.x] += rs[threadIdx.x + o]; rq[threadIdx.x] += rq[threadIdx.x + o]; }
        __syncthreads();
    }
    float mean = rs[0] / (float)DD;
    float var  = rq[0] / (float)DD - mean * mean;
    float inv  = rsqrtf(var + 1e-6f);
    #pragma unroll
    for (int i = 0; i < DD / 256; i++) {
        int d = threadIdx.x + i * 256;
        out16[(size_t)row * DD + d] = f2bf(w[d] * (vals[i] - mean) * inv);
    }
}

// ---------------- scalars: ponder, ekf ----------------
__global__ void
finalize_kernel(const float* __restrict__ ek, float* __restrict__ out, long off) {
    if (blockIdx.x == 0 && threadIdx.x == 0) {
        float p = (ek[0] + ek[1] + ek[2]) / 3.f;
        float e = (fmaxf(-ek[0], 0.f) + fmaxf(-ek[1], 0.f) + fmaxf(-ek[2], 0.f)) / 3.f;
        out[off] = p;
        out[off + 1] = e;
    }
}

// ==================== host launch ====================
extern "C" void kernel_launch(void* const* d_in, const int* in_sizes, int n_in,
                              void* d_out, int out_size, void* d_ws, size_t ws_size,
                              hipStream_t stream) {
    const int*   tok        = (const int*)  d_in[0];
    const float* embed      = (const float*)d_in[1];
    const float* norm_w     = (const float*)d_in[2];
    const float* decode_W   = (const float*)d_in[3];
    const float* decode_b   = (const float*)d_in[4];
    const float* out_norm_w = (const float*)d_in[5];
    const float* beta_out   = (const float*)d_in[6];
    const float* vth_out    = (const float*)d_in[7];
    const float* H_logits   = (const float*)d_in[8];
    const float* w_pre      = (const float*)d_in[9];
    const float* w_post     = (const float*)d_in[10];
    const float* beta_in1   = (const float*)d_in[11];
    const float* vth_in1    = (const float*)d_in[12];
    const float* W_in       = (const float*)d_in[13];
    const float* beta_h     = (const float*)d_in[14];
    const float* vth_h      = (const float*)d_in[15];
    const float* W_out      = (const float*)d_in[16];
    const float* beta_in2   = (const float*)d_in[17];
    const float* vth_in2    = (const float*)d_in[18];
    const float* W_gate     = (const float*)d_in[19];
    const float* beta_gate  = (const float*)d_in[20];
    const float* vth_gate   = (const float*)d_in[21];
    const float* W_up       = (const float*)d_in[22];
    const float* beta_up    = (const float*)d_in[23];
    const float* vth_up     = (const float*)d_in[24];
    const float* W_down     = (const float*)d_in[25];
    const float* halt_w     = (const float*)d_in[26];
    const float* halt_b     = (const float*)d_in[27];

    float* ws = (float*)d_ws;
    size_t o = 0;
    auto alloc = [&](size_t n) { size_t r = o; o += ((n + 15) & ~(size_t)15); return r; };
    const size_t SZ_H  = (size_t)SS * BB * NST * DD;   // 2,097,152
    const size_t SZ_X  = (size_t)SS * BB * DD;         //   524,288
    const size_t SZ_TD = (size_t)TT * BB * DD;         // 2,097,152
    const size_t SZ_TN = (size_t)TT * BB * NDD;        // 8,388,608
    const size_t SZ_TF = (size_t)TT * BB * DFF;        // 6,291,456

    // fp32 buffers
    float* h     = ws + alloc(SZ_H);
    float* hmix  = ws + alloc(SZ_H);
    float* x     = ws + alloc(SZ_X);
    float* big   = ws + alloc(SZ_TN);      // s1@W_in gemm out (fp32)
    float* yblk  = ws + alloc(SZ_TD);
    float* gatef = ws + alloc(SZ_TF);      // gate gemm out
    float* upf   = ws + alloc(SZ_TF);      // up gemm out; later reused as y (fp32)
    float* hn    = ws + alloc(SZ_X);
    float* hhf   = ws + alloc(SZ_X);
    float* lam   = ws + alloc((size_t)SS * KK * BB);
    float* pk    = ws + alloc((size_t)SS * BB * KK);
    float* Hs    = ws + alloc(LL * 16);
    float* wpre  = ws + alloc(LL * 4);
    float* wpost = ws + alloc(LL * 4);
    float* ek    = ws + alloc(16);
    // bf16 buffers (ushort)
    unsigned short* s1_16 = (unsigned short*)(ws + alloc((SZ_TD + 1) / 2));
    unsigned short* sh_16 = (unsigned short*)(ws + alloc((SZ_TN + 1) / 2));
    unsigned short* s2_16 = (unsigned short*)(ws + alloc((SZ_TD + 1) / 2));
    unsigned short* gu_16 = (unsigned short*)(ws + alloc((SZ_TF + 1) / 2));
    unsigned short* dec16 = (unsigned short*)(ws + alloc((SZ_X + 1) / 2));
    unsigned short* hh16  = (unsigned short*)(ws + alloc((SZ_X + 1) / 2));
    unsigned short* wbuf  = (unsigned short*)(ws + alloc(((size_t)DD * NDD + 1) / 2));
    unsigned short* emb16 = (unsigned short*)(ws + alloc(((size_t)VV * DD + 1) / 2));
    unsigned short* dw16  = (unsigned short*)(ws + alloc(((size_t)DD * DD + 1) / 2));

    auto blocks = [](size_t n) { return dim3((unsigned)((n + 255) / 256)); };
    auto gemm = [&](int epi, const unsigned short* A, const unsigned short* Bt, float* C,
                    const float* aux, int M, int N, int K) {
        size_t waves = (size_t)(M >> 5) * (N >> 6);
        dim3 g = blocks(waves * 32);
        if (epi == 1)
            gemm_bf16_kernel<1><<<g, 256, 0, stream>>>(A, Bt, C, aux, M, N, K);
        else if (epi == 2)
            gemm_bf16_kernel<2><<<g, 256, 0, stream>>>(A, Bt, C, aux, M, N, K);
        else
            gemm_bf16_kernel<0><<<g, 256, 0, stream>>>(A, Bt, C, nullptr, M, N, K);
    };

    // params (sinkhorn/softmax) + ek zeroing, embedding init, static bf16 weights
    params_kernel<<<LL, 32, 0, stream>>>(H_logits, w_pre, w_post, Hs, wpre, wpost, ek);
    embed_init_kernel<<<blocks((size_t)SS * BB * DD), 256, 0, stream>>>(tok, embed, h);
    convC_kernel<<<blocks((size_t)VV * DD), 256, 0, stream>>>(embed, emb16, (long)VV * DD);
    convC_kernel<<<blocks((size_t)DD * DD), 256, 0, stream>>>(decode_W, dw16, (long)DD * DD);

    const int Mrows = TT * BB;   // 2048
    for (int l = 0; l < LL; l++) {
        mix_kernel<<<blocks((size_t)SS * BB * DD), 256, 0, stream>>>(h, Hs + l * 16, wpre + l * 4, hmix, x);
        plif_kernel<<<blocks((size_t)BB * DD), 256, 0, stream>>>(
            x, 1, nullptr, beta_in1 + l * DD, vth_in1 + l * DD, s1_16, TT, BB, DD);

        convT_kernel<<<blocks((size_t)DD * NDD), 256, 0, stream>>>(W_in + (size_t)l * DD * NDD, wbuf, DD, NDD);
        gemm(0, s1_16, wbuf, big, nullptr, Mrows, NDD, DD);
        plif_kernel<<<blocks((size_t)BB * NDD), 256, 0, stream>>>(
            big, 0, nullptr, beta_h + l * NDD, vth_h + l * NDD, sh_16, TT, BB, NDD);

        convT_kernel<<<blocks((size_t)NDD * DD), 256, 0, stream>>>(W_out + (size_t)l * NDD * DD, wbuf, NDD, DD);
        gemm(0, sh_16, wbuf, yblk, nullptr, Mrows, DD, NDD);

        plif_kernel<<<blocks((size_t)BB * DD), 256, 0, stream>>>(
            x, 1, yblk, beta_in2 + l * DD, vth_in2 + l * DD, s2_16, TT, BB, DD);

        convT_kernel<<<blocks((size_t)DD * DFF), 256, 0, stream>>>(W_gate + (size_t)l * DD * DFF, wbuf, DD, DFF);
        gemm(0, s2_16, wbuf, gatef, nullptr, Mrows, DFF, DD);
        convT_kernel<<<blocks((size_t)DD * DFF), 256, 0, stream>>>(W_up + (size_t)l * DD * DFF, wbuf, DD, DFF);
        gemm(0, s2_16, wbuf, upf, nullptr, Mrows, DFF, DD);

        gateup_kernel<<<blocks((size_t)BB * DFF), 256, 0, stream>>>(
            gatef, upf, beta_gate + l * DFF, vth_gate + l * DFF,
            beta_up + l * DFF, vth_up + l * DFF, gu_16, TT, BB, DFF);

        convT_kernel<<<blocks((size_t)DFF * DD), 256, 0, stream>>>(W_down + (size_t)l * DFF * DD, wbuf, DFF, DD);
        float* y = upf;  // reuse (up gemm out is dead after gateup)
        gemm(1, gu_16, wbuf, y, yblk, Mrows, DD, DFF);

        halt_lam_kernel<<<blocks((size_t)Mrows * 32), 256, 0, stream>>>(y, halt_w + l * DD, halt_b + l, lam, Mrows);
        pk_ek_kernel<<<blocks((size_t)SS * BB), 256, 0, stream>>>(lam, pk, ek + l);
        ytok_update_kernel<<<blocks((size_t)SS * BB * DD), 256, 0, stream>>>(y, pk, hmix, wpost + l * 4, h);
    }

    // decode path
    collapse_rms_kernel<<<SS * BB, 256, 0, stream>>>(h, out_norm_w, hn);
    plif_out_kernel<<<blocks((size_t)BB * DD), 256, 0, stream>>>(hn, beta_out, vth_out, dec16);
    gemm(2, dec16, dw16, hhf, decode_b, BB * SS, DD, DD);
    lateral_kernel<<<BB * SS, 256, 0, stream>>>(hhf, norm_w, hh16);
    gemm(0, hh16, emb16, (float*)d_out, nullptr, BB * SS, VV, DD);

    finalize_kernel<<<1, 32, 0, stream>>>(ek, (float*)d_out, (long)BB * SS * VV);
}
